// CapsNet_73650099192354
// MI455X (gfx1250) — matrix operand order repaired and checked by reading, not compiled, not run on hardware
//
#include <hip/hip_runtime.h>
#include <cstdint>
#include <cstddef>

// ---------------------------------------------------------------------------
// CDNA5 WMMA types/helpers (wave32, 16x16x32 bf16 -> f32)
// ---------------------------------------------------------------------------
typedef __attribute__((ext_vector_type(16))) __bf16 v16bf;
typedef __attribute__((ext_vector_type(8)))  float  v8f;

union Frag   { v16bf v; uint4 q[2]; };
union PackBF { uint4 q; __bf16 e[8]; };

__device__ inline void frag_zero(Frag& f) {
  f.q[0] = make_uint4(0u, 0u, 0u, 0u);
  f.q[1] = make_uint4(0u, 0u, 0u, 0u);
}

__device__ inline v8f wmma_bf16(const Frag& a, const Frag& b, v8f c) {
  // 8 args: (neg_a, A, neg_b, B, c_mod, C, reuse_a, reuse_b)
  return __builtin_amdgcn_wmma_f32_16x16x32_bf16(false, a.v, false, b.v,
                                                 (short)0, c, false, false);
}

// ---------------------------------------------------------------------------
// conv1: C_in=1 (K=9 too small for WMMA) — direct VALU conv + BN + ReLU
// x: [32][224][224] f32 (NCHW, C=1);  out: NHWC bf16 [pix][32]
// ---------------------------------------------------------------------------
__global__ __launch_bounds__(256) void conv1_bn_relu(
    const float* __restrict__ x, const float* __restrict__ w1,
    const float* __restrict__ bn, __bf16* __restrict__ out, int Mtot) {
  int pix = blockIdx.x * 256 + threadIdx.x;
  if (pix >= Mtot) return;
  const int HW = 224 * 224;
  int b = pix / HW, rem = pix % HW, y = rem / 224, xx = rem % 224;
  const float* xb = x + (size_t)b * HW;
  float patch[9];
#pragma unroll
  for (int t = 0; t < 9; ++t) {
    int iy = y + t / 3 - 1, ix = xx + t % 3 - 1;
    patch[t] = (iy >= 0 && iy < 224 && ix >= 0 && ix < 224) ? xb[iy * 224 + ix] : 0.f;
  }
#pragma unroll
  for (int o = 0; o < 32; ++o) {
    float s = 0.f;
#pragma unroll
    for (int t = 0; t < 9; ++t) s += patch[t] * w1[o * 9 + t];
    float sc = bn[o] * rsqrtf(bn[96 + o] + 1e-5f);
    float sh = bn[32 + o] - bn[64 + o] * sc;
    out[(size_t)pix * 32 + o] = (__bf16)fmaxf(s * sc + sh, 0.f);
  }
}

// ---------------------------------------------------------------------------
// Weight prep: OIHW f32 -> [tap][O][C] bf16 (K-chunks contiguous in ci)
// ---------------------------------------------------------------------------
__global__ __launch_bounds__(256) void prep_conv_w(
    const float* __restrict__ w, __bf16* __restrict__ wp, int O, int C) {
  int idx = blockIdx.x * 256 + threadIdx.x;
  int total = 9 * O * C;
  if (idx >= total) return;
  int t = idx / (O * C), rem = idx % (O * C), o = rem / C, c = rem % C;
  wp[idx] = (__bf16)w[(size_t)(o * C + c) * 9 + t];
}

// ---------------------------------------------------------------------------
// 3x3 SAME conv as implicit GEMM via WMMA.
//   A = weights  [M=COUT rows x K=CIN*9], staged once per block in LDS
//   B = im2col   [K x N=16 pixels]       (NHWC activations, on-the-fly)
//   D lane layout: row o = j + 8*(lane>>4), col pixel = lane&15
//   -> epilogue: each lane stores 8 consecutive bf16 channels (16B store)
// Block = 128 threads = 4 waves; each wave = TWO 16-pixel tiles (A fragments
// reused across both B tiles -> 2x wmma per weight fetch), full COUT.
// ---------------------------------------------------------------------------
template <int CIN, int COUT, int H, int W>
__global__ __launch_bounds__(128) void conv3x3_wmma(
    const __bf16* __restrict__ in,   // [B][H][W][CIN] NHWC bf16
    const __bf16* __restrict__ wp,   // [9][COUT][CIN] bf16
    const float* __restrict__ bn,    // [4][COUT]
    __bf16* __restrict__ out,        // [B][H][W][COUT]
    int Mtot) {
  constexpr int WELEMS = 9 * COUT * CIN;     // <= 36864 elems (72KB) -> fits LDS
  __shared__ __bf16 wlds[WELEMS];
  {
    const uint4* src = (const uint4*)wp;
    uint4* dst = (uint4*)wlds;
    for (int i = threadIdx.x; i < WELEMS / 8; i += 128) dst[i] = src[i];
  }
  __syncthreads();

  const int lane = threadIdx.x & 31;
  const int wave = threadIdx.x >> 5;
  const int h    = lane >> 4;                 // lane-half selects K sub-block
  const int ml   = lane & 15;
  const int tileN = (blockIdx.x * 4 + wave) * 32;   // 2 pixel tiles per wave
  constexpr int MT = COUT / 16;               // channel tiles

  int pb[2], py[2], px[2], pix[2];
  bool pv[2];
#pragma unroll
  for (int n2 = 0; n2 < 2; ++n2) {
    pix[n2] = tileN + n2 * 16 + ml;
    pv[n2] = pix[n2] < Mtot;
    pb[n2] = py[n2] = px[n2] = 0;
    if (pv[n2]) {
      pb[n2] = pix[n2] / (H * W);
      int r = pix[n2] % (H * W);
      py[n2] = r / W; px[n2] = r % W;
    }
  }

  v8f acc[MT][2];
#pragma unroll
  for (int mt = 0; mt < MT; ++mt)
#pragma unroll
    for (int n2 = 0; n2 < 2; ++n2)
#pragma unroll
      for (int j = 0; j < 8; ++j) acc[mt][n2][j] = 0.f;

#pragma unroll
  for (int t = 0; t < 9; ++t) {
    // prefetch next tap's activation row into the cache hierarchy
    if (t < 8) {
      int iy2 = py[0] + (t + 1) / 3 - 1, ix2 = px[0] + (t + 1) % 3 - 1;
      if (pv[0] && iy2 >= 0 && iy2 < H && ix2 >= 0 && ix2 < W)
        __builtin_prefetch(in + ((size_t)(pb[0] * H + iy2) * W + ix2) * CIN, 0, 1);
    }
    int iy[2], ix[2];
    bool v[2];
#pragma unroll
    for (int n2 = 0; n2 < 2; ++n2) {
      iy[n2] = py[n2] + t / 3 - 1;
      ix[n2] = px[n2] + t % 3 - 1;
      v[n2] = pv[n2] && iy[n2] >= 0 && iy[n2] < H && ix[n2] >= 0 && ix[n2] < W;
    }
#pragma unroll
    for (int cb = 0; cb < CIN; cb += 32) {
      // B fragments: col = pixel, rows ci = cb + h*16 .. +15 (one 32B load)
      Frag bfr[2];
#pragma unroll
      for (int n2 = 0; n2 < 2; ++n2) {
        if (v[n2]) {
          const __bf16* p =
              in + ((size_t)(pb[n2] * H + iy[n2]) * W + ix[n2]) * CIN + cb + h * 16;
          bfr[n2].q[0] = *(const uint4*)(p);
          bfr[n2].q[1] = *(const uint4*)(p + 8);
        } else {
          frag_zero(bfr[n2]);
        }
      }
#pragma unroll
      for (int mt = 0; mt < MT; ++mt) {
        // A fragment from LDS: row o = mt*16+ml; K = cb+h*16 .. +15 contiguous
        Frag af;
        const __bf16* q = wlds + ((size_t)t * COUT + (mt * 16 + ml)) * CIN + cb;
        af.q[0] = *(const uint4*)(q + h * 16);
        af.q[1] = *(const uint4*)(q + h * 16 + 8);
        acc[mt][0] = wmma_bf16(af, bfr[0], acc[mt][0]);
        acc[mt][1] = wmma_bf16(af, bfr[1], acc[mt][1]);
      }
    }
  }

#pragma unroll
  for (int n2 = 0; n2 < 2; ++n2) {
    if (!pv[n2]) continue;
#pragma unroll
    for (int mt = 0; mt < MT; ++mt) {
      const int obase = mt * 16 + h * 8;
      PackBF pk;
#pragma unroll
      for (int j = 0; j < 8; ++j) {
        const int o = obase + j;
        float sc = bn[o] * rsqrtf(bn[3 * COUT + o] + 1e-5f);
        float sh = bn[COUT + o] - bn[2 * COUT + o] * sc;
        pk.e[j] = (__bf16)fmaxf(acc[mt][n2][j] * sc + sh, 0.f);
      }
      *(uint4*)(out + (size_t)pix[n2] * COUT + obase) = pk.q;
    }
  }
}

// ---------------------------------------------------------------------------
// avgpool 2x2 on NHWC bf16
// ---------------------------------------------------------------------------
__global__ __launch_bounds__(256) void avgpool2_nhwc(
    const __bf16* __restrict__ in, __bf16* __restrict__ out,
    int B, int H, int W, int C) {
  int idx = blockIdx.x * 256 + threadIdx.x;
  int total = B * (H / 2) * (W / 2) * C;
  if (idx >= total) return;
  int c = idx % C, r = idx / C;
  int ox = r % (W / 2); r /= (W / 2);
  int oy = r % (H / 2); int b = r / (H / 2);
  size_t base = ((size_t)(b * H + oy * 2) * W + ox * 2) * C + c;
  float s = (float)in[base] + (float)in[base + C] +
            (float)in[base + (size_t)W * C] + (float)in[base + (size_t)W * C + C];
  out[idx] = (__bf16)(s * 0.25f);
}

// ---------------------------------------------------------------------------
// route_weights transpose+convert: [r][c][n] f32 -> [r][n][c] bf16  (n=o*8+d)
// ---------------------------------------------------------------------------
__global__ __launch_bounds__(256) void prep_rw(
    const float* __restrict__ rw, __bf16* __restrict__ rwT) {
  int idx = blockIdx.x * 256 + threadIdx.x;
  if (idx >= 3136 * 200 * 64) return;
  int r = idx / (200 * 64), rem = idx % (200 * 64), n = rem / 64, c = rem % 64;
  rwT[idx] = (__bf16)rw[((size_t)r * 64 + c) * 200 + n];
}

// ---------------------------------------------------------------------------
// priors[b][r][n] = sum_c feats[b][r][c] * rwT[r][n][c]   (per-r GEMM)
//   A = rwT rows n (M=200 -> 13 tiles, grid.y), B = feats cols b (2 waves)
// ---------------------------------------------------------------------------
__global__ __launch_bounds__(64) void priors_wmma(
    const __bf16* __restrict__ feats,  // [32][3136][64]
    const __bf16* __restrict__ rwT,    // [3136][200][64]
    float* __restrict__ priors) {      // [32][3136][200]
  const int r  = blockIdx.x;
  const int nt = blockIdx.y;             // 0..12
  const int lane = threadIdx.x & 31;
  const int btile = threadIdx.x >> 5;    // 0/1
  const int h  = lane >> 4;
  const int ml = lane & 15;
  const int bidx = btile * 16 + ml;      // batch column (<32 always)
  const int nrow = nt * 16 + ml;         // A row

  v8f acc;
#pragma unroll
  for (int j = 0; j < 8; ++j) acc[j] = 0.f;

#pragma unroll
  for (int ks = 0; ks < 2; ++ks) {
    Frag bf;  // feats column bidx, rows c = ks*32 + h*16 .. +15
    const __bf16* p = feats + ((size_t)bidx * 3136 + r) * 64 + ks * 32 + h * 16;
    bf.q[0] = *(const uint4*)(p);
    bf.q[1] = *(const uint4*)(p + 8);
    Frag af;  // rwT row nrow, K = ks*32 + h*16 .. +15 contiguous
    if (nrow < 200) {
      const __bf16* q = rwT + ((size_t)r * 200 + nrow) * 64 + ks * 32;
      af.q[0] = *(const uint4*)(q + h * 16);
      af.q[1] = *(const uint4*)(q + h * 16 + 8);
    } else {
      frag_zero(af);
    }
    acc = wmma_bf16(af, bf, acc);
  }

  const int nbase = nt * 16 + h * 8;     // 8 consecutive n per lane
  if (nbase + 8 <= 200) {
    float* dst = priors + ((size_t)bidx * 3136 + r) * 200 + nbase;
    *(float4*)(dst)     = make_float4(acc[0], acc[1], acc[2], acc[3]);
    *(float4*)(dst + 4) = make_float4(acc[4], acc[5], acc[6], acc[7]);
  }
}

// ---------------------------------------------------------------------------
// Routing: softmax over R per (b,o)
// ---------------------------------------------------------------------------
__global__ __launch_bounds__(256) void softmax_r(
    const float* __restrict__ logits, float* __restrict__ probs) {
  const int b = blockIdx.x / 25, o = blockIdx.x % 25;
  const size_t base = (size_t)b * 3136 * 25 + o;
  const int tid = threadIdx.x;
  __shared__ float red[256];

  float mx = -3.0e38f;
  for (int r = tid; r < 3136; r += 256)
    mx = fmaxf(mx, logits[base + (size_t)r * 25]);
  red[tid] = mx; __syncthreads();
  for (int s = 128; s > 0; s >>= 1) {
    if (tid < s) red[tid] = fmaxf(red[tid], red[tid + s]);
    __syncthreads();
  }
  mx = red[0]; __syncthreads();

  float sum = 0.f;
  for (int r = tid; r < 3136; r += 256)
    sum += __expf(logits[base + (size_t)r * 25] - mx);
  red[tid] = sum; __syncthreads();
  for (int s = 128; s > 0; s >>= 1) {
    if (tid < s) red[tid] += red[tid + s];
    __syncthreads();
  }
  float inv = 1.f / red[0];

  for (int r = tid; r < 3136; r += 256)
    probs[base + (size_t)r * 25] = __expf(logits[base + (size_t)r * 25] - mx) * inv;
}

// ---------------------------------------------------------------------------
// s[b][o][:] = sum_r probs[b][r][o] * priors[b][r][o][:]; then squash -> v
// ---------------------------------------------------------------------------
__global__ __launch_bounds__(256) void s_squash(
    const float* __restrict__ probs, const float* __restrict__ priors,
    float* __restrict__ vout) {
  const int b = blockIdx.x / 25, o = blockIdx.x % 25;
  const int tid = threadIdx.x;
  float s[8];
#pragma unroll
  for (int d = 0; d < 8; ++d) s[d] = 0.f;

  for (int r = tid; r < 3136; r += 256) {
    float p = probs[((size_t)b * 3136 + r) * 25 + o];
    const float* pr = priors + ((size_t)b * 3136 + r) * 200 + o * 8;
#pragma unroll
    for (int d = 0; d < 8; ++d) s[d] += p * pr[d];
  }

  __shared__ float red[8 * 256];
#pragma unroll
  for (int d = 0; d < 8; ++d) red[d * 256 + tid] = s[d];
  __syncthreads();
  for (int st = 128; st > 0; st >>= 1) {
    if (tid < st)
#pragma unroll
      for (int d = 0; d < 8; ++d) red[d * 256 + tid] += red[d * 256 + tid + st];
    __syncthreads();
  }
  if (tid == 0) {
    float sv[8], n2 = 0.f;
#pragma unroll
    for (int d = 0; d < 8; ++d) { sv[d] = red[d * 256]; n2 += sv[d] * sv[d]; }
    float nn = sqrtf(n2);
    float f = (n2 / (1.f + n2)) / fmaxf(nn, 1e-8f);
#pragma unroll
    for (int d = 0; d < 8; ++d)
      vout[((size_t)b * 25 + o) * 8 + d] = sv[d] * f;
  }
}

// ---------------------------------------------------------------------------
// logits[b][r][o] += dot(priors[b][r][o][:], v[b][o][:])
// ---------------------------------------------------------------------------
__global__ __launch_bounds__(256) void update_logits(
    const float* __restrict__ priors, const float* __restrict__ v,
    float* __restrict__ logits, int total) {
  int idx = blockIdx.x * 256 + threadIdx.x;
  if (idx >= total) return;
  int o = idx % 25, rem = idx / 25, r = rem % 3136, b = rem / 3136;
  const float* pr = priors + ((size_t)b * 3136 + r) * 200 + o * 8;
  const float* vv = v + ((size_t)b * 25 + o) * 8;
  float d = 0.f;
#pragma unroll
  for (int k = 0; k < 8; ++k) d += pr[k] * vv[k];
  logits[idx] += d;
}

__global__ __launch_bounds__(256) void zero_f32(float* __restrict__ p, int n) {
  int idx = blockIdx.x * 256 + threadIdx.x;
  if (idx < n) p[idx] = 0.f;
}

// ---------------------------------------------------------------------------
// FC: out[b][j] = caps[b][:] . fc_w[j][:] + fc_b[j]
// ---------------------------------------------------------------------------
__global__ __launch_bounds__(64) void fc_kernel(
    const float* __restrict__ caps, const float* __restrict__ w,
    const float* __restrict__ bias, float* __restrict__ out) {
  int idx = threadIdx.x;
  if (idx >= 64) return;
  int b = idx >> 1, j = idx & 1;
  float s = bias[j];
  for (int k = 0; k < 200; ++k) s += caps[b * 200 + k] * w[j * 200 + k];
  out[idx] = s;
}

// ---------------------------------------------------------------------------
// Host launcher
// ---------------------------------------------------------------------------
extern "C" void kernel_launch(void* const* d_in, const int* in_sizes, int n_in,
                              void* d_out, int out_size, void* d_ws, size_t ws_size,
                              hipStream_t stream) {
  const float* x    = (const float*)d_in[0];
  const float* w1   = (const float*)d_in[1];
  const float* bn1  = (const float*)d_in[2];
  const float* w2   = (const float*)d_in[3];
  const float* bn2  = (const float*)d_in[4];
  const float* w3   = (const float*)d_in[5];
  const float* bn3  = (const float*)d_in[6];
  const float* w4   = (const float*)d_in[7];
  const float* bn4  = (const float*)d_in[8];
  const float* rw   = (const float*)d_in[9];
  const float* fcw  = (const float*)d_in[10];
  const float* fcb  = (const float*)d_in[11];
  float* out = (float*)d_out;

  char* ws = (char*)d_ws;
  size_t off = 0;
  auto take = [&](size_t bytes) -> char* {
    char* p = ws + off;
    off += (bytes + 255) & ~(size_t)255;
    return p;
  };

  const size_t ACT1  = (size_t)32 * 224 * 224 * 32 * 2;  // also holds act4 later
  const size_t ACT2  = (size_t)32 * 224 * 224 * 32 * 2;  // also holds rwT later
  const size_t POOLC = (size_t)32 * 112 * 112 * 32 * 2;  // also holds feats later
  const size_t ACT3S = (size_t)32 * 112 * 112 * 64 * 2;
  const size_t PRI   = (size_t)32 * 3136 * 200 * 4;
  const size_t LOG   = (size_t)32 * 3136 * 25 * 4;

  char* bufA = take(ACT1);            // act1, then act4
  char* bufB = take(ACT2);            // act2, then rwT (80.3MB <= 102.8MB)
  char* bufC = take(POOLC);           // pool1, then feats (12.8MB <= 25.7MB)
  char* bufD = take(ACT3S);           // act3
  float* priors = (float*)take(PRI);
  float* logits = (float*)take(LOG);
  float* probs  = (float*)take(LOG);
  float* vcaps  = (float*)take((size_t)32 * 200 * 4);
  __bf16* w2p = (__bf16*)take((size_t)9 * 32 * 32 * 2);
  __bf16* w3p = (__bf16*)take((size_t)9 * 64 * 32 * 2);
  __bf16* w4p = (__bf16*)take((size_t)9 * 64 * 64 * 2);

  __bf16* act1  = (__bf16*)bufA;
  __bf16* act4  = (__bf16*)bufA;
  __bf16* act2  = (__bf16*)bufB;
  __bf16* rwT   = (__bf16*)bufB;
  __bf16* pool1 = (__bf16*)bufC;
  __bf16* feats = (__bf16*)bufC;
  __bf16* act3  = (__bf16*)bufD;

  // ---- conv stack ----
  const int M1 = 32 * 224 * 224;   // pixels at 224x224
  const int M2 = 32 * 112 * 112;   // pixels at 112x112
  conv1_bn_relu<<<(M1 + 255) / 256, 256, 0, stream>>>(x, w1, bn1, act1, M1);

  prep_conv_w<<<(9 * 32 * 32 + 255) / 256, 256, 0, stream>>>(w2, w2p, 32, 32);
  conv3x3_wmma<32, 32, 224, 224><<<M1 / 128, 128, 0, stream>>>(act1, w2p, bn2, act2, M1);

  avgpool2_nhwc<<<(M2 * 32 + 255) / 256, 256, 0, stream>>>(act2, pool1, 32, 224, 224, 32);

  prep_conv_w<<<(9 * 64 * 32 + 255) / 256, 256, 0, stream>>>(w3, w3p, 64, 32);
  conv3x3_wmma<32, 64, 112, 112><<<M2 / 128, 128, 0, stream>>>(pool1, w3p, bn3, act3, M2);

  prep_conv_w<<<(9 * 64 * 64 + 255) / 256, 256, 0, stream>>>(w4, w4p, 64, 64);
  conv3x3_wmma<64, 64, 112, 112><<<M2 / 128, 128, 0, stream>>>(act3, w4p, bn4, act4, M2);

  const int MF = 32 * 56 * 56 * 64;  // feats element count
  avgpool2_nhwc<<<(MF + 255) / 256, 256, 0, stream>>>(act4, feats, 32, 112, 112, 64);

  // ---- capsule priors (batched WMMA GEMM per r) ----
  prep_rw<<<(3136 * 200 * 64 + 255) / 256, 256, 0, stream>>>(rw, rwT);
  priors_wmma<<<dim3(3136, 13), 64, 0, stream>>>(feats, rwT, priors);

  // ---- dynamic routing (3 iterations) ----
  const int NL = 32 * 3136 * 25;
  zero_f32<<<(NL + 255) / 256, 256, 0, stream>>>(logits, NL);
  for (int it = 0; it < 3; ++it) {
    softmax_r<<<32 * 25, 256, 0, stream>>>(logits, probs);
    s_squash<<<32 * 25, 256, 0, stream>>>(probs, priors, vcaps);
    if (it < 2)
      update_logits<<<(NL + 255) / 256, 256, 0, stream>>>(priors, vcaps, logits, NL);
  }

  // ---- final FC ----
  fc_kernel<<<1, 64, 0, stream>>>(vcaps, fcw, fcb, out);

  (void)in_sizes; (void)n_in; (void)out_size; (void)ws_size;
}
